// MultiHeadAttention_2001454760342
// MI455X (gfx1250) — matrix operand order repaired and checked by reading, not compiled, not run on hardware
//
#include <hip/hip_runtime.h>

// ---------------------------------------------------------------------------
// MI455X (gfx1250) multi-head attention.
// GEMMs: v_wmma_f32_16x16x32_bf16, 32x64 C tile per wave (8 wmma / K-step),
// A and B tiles staged in LDS; bf16 tiles via GLOBAL_LOAD_ASYNC_TO_LDS_B128,
// f32 A (attn) converted once per block via v_cvt (convertvector) + ds_store.
// Softmax in f32. attn matrix is a required output -> materialized in d_out.
// ---------------------------------------------------------------------------

typedef unsigned short u16;
typedef __attribute__((ext_vector_type(16))) __bf16 v16bf;
typedef __attribute__((ext_vector_type(4)))  __bf16 v4bf;
typedef __attribute__((ext_vector_type(8)))  float  v8f;
typedef __attribute__((ext_vector_type(4)))  unsigned int uv4;
typedef __attribute__((ext_vector_type(4)))  float  fv4;
typedef __attribute__((ext_vector_type(4)))  int    iv4;

#ifndef __has_builtin
#define __has_builtin(x) 0
#endif

#if __has_builtin(__builtin_amdgcn_global_load_async_to_lds_b128)
#define HAVE_ASYNC_LDS 1
typedef __attribute__((address_space(1))) iv4 iv4_g;   // global (AS1) int4
typedef __attribute__((address_space(3))) iv4 iv4_l;   // LDS (AS3) int4
#define ASYNC_CP16(src, dst) \
  __builtin_amdgcn_global_load_async_to_lds_b128( \
      (iv4_g*)(void*)(const void*)(src), (iv4_l*)(void*)(dst), 0, 0)
#if __has_builtin(__builtin_amdgcn_s_wait_asynccnt)
#define WAIT_ASYNC() __builtin_amdgcn_s_wait_asynccnt(0)
#else
#define WAIT_ASYNC() asm volatile("s_wait_asynccnt 0x0" ::: "memory")
#endif
#else
#define HAVE_ASYNC_LDS 0
#define ASYNC_CP16(src, dst) (*(uv4*)(dst) = *(const uv4*)(src))
#define WAIT_ASYNC()
#endif

static __device__ __forceinline__ u16 f32_to_bf16(float x) {
  unsigned u = __float_as_uint(x);
  u += 0x7FFFu + ((u >> 16) & 1u);   // round-to-nearest-even
  return (u16)(u >> 16);
}

union BFrag {
  uv4   q[2];
  u16   s[16];
  v16bf v;
};

static __device__ __forceinline__ v8f wmma_bf16(const BFrag& a, const BFrag& b, v8f c) {
  return __builtin_amdgcn_wmma_f32_16x16x32_bf16(false, a.v, false, b.v,
                                                 (short)0, c, false, false);
}

// C[M,N] = alpha * A[M,K] x B[K,N], B supplied transposed as Bt[N,K] (bf16).
// A: bf16 (aIsF32=0) or f32 converted once per block into LDS (aIsF32=1).
// cMode: 0 = f32 store, 1 = bf16 store, 2 = bf16 store transposed (C[n,m]).
// Block = 256 threads = 8 waves (4 M x 2 N); block tile 128x128;
// wave tile 32(M) x 64(N) = 8 accumulators, 8 wmma per K-step.
// grid = (N/128, M/128, Z).
__global__ __launch_bounds__(256) void gemm_wmma_bf16(
    const void* __restrict__ Aptr, int lda, long long aOffZ, int aIsF32,
    const u16*  __restrict__ Bptr, int ldb, long long bOffZ,
    void*       __restrict__ Cptr, int ldc, long long cOffZ, int cMode,
    int Kdim, float alpha)
{
  __shared__ u16 Bs[128 * 32];     // 8 KB: B block tile [n:128][k:32]
  __shared__ u16 As[128 * 32];     // 8 KB: A block tile [m:128][k:32]

  const int tid  = threadIdx.x;
  const int lane = tid & 31;
  const int wave = tid >> 5;
  const int hh   = lane >> 4;      // lane half
  const int ln   = lane & 15;
  const int wm   = wave & 3;       // 4 M sub-tiles
  const int wn   = wave >> 2;      // 2 N sub-tiles
  const int mblk = blockIdx.y * 128;
  const int nblk = blockIdx.x * 128;
  const int m0   = mblk + wm * 32;
  const int nloc = wn * 64;
  const int n0   = nblk + nloc;
  const int z    = blockIdx.z;

  const v8f vzero = {0.f, 0.f, 0.f, 0.f, 0.f, 0.f, 0.f, 0.f};
  v8f acc[2][4] = {{vzero, vzero, vzero, vzero}, {vzero, vzero, vzero, vzero}};

  const u16*   Bg  = Bptr + bOffZ * z + (long long)nblk * ldb;
  const u16*   Ag  = (const u16*)Aptr + aOffZ * z + (long long)mblk * lda;
  const float* Agf = (const float*)Aptr + aOffZ * z + (long long)mblk * lda;

  for (int k0 = 0; k0 < Kdim; k0 += 32) {
    // ---- stage B[nblk..+127][k0..+31]: 512 x 16B chunks, async ----
#pragma unroll
    for (int it = 0; it < 2; ++it) {
      const int c = tid + it * 256;
      ASYNC_CP16(Bg + (long long)(c >> 2) * ldb + k0 + (c & 3) * 8, &Bs[c * 8]);
    }
    // ---- stage A[mblk..+127][k0..+31] ----
    if (aIsF32) {
#pragma unroll
      for (int it = 0; it < 2; ++it) {
        const int c = tid + it * 256;
        const float* s = Agf + (long long)(c >> 2) * lda + k0 + (c & 3) * 8;
        fv4 f0 = *(const fv4*)s;
        fv4 f1 = *(const fv4*)(s + 4);
        union { v4bf h[2]; uv4 q; } pk;
        pk.h[0] = __builtin_convertvector(f0, v4bf);   // v_cvt f32->bf16
        pk.h[1] = __builtin_convertvector(f1, v4bf);
        *(uv4*)&As[c * 8] = pk.q;
      }
    } else {
#pragma unroll
      for (int it = 0; it < 2; ++it) {
        const int c = tid + it * 256;
        ASYNC_CP16(Ag + (long long)(c >> 2) * lda + k0 + (c & 3) * 8, &As[c * 8]);
      }
    }
    WAIT_ASYNC();
    __syncthreads();

    // ---- fragments from LDS ----
    // A lane layout: K(i) = 16*(i>>3) + 8*hh + (i&7)
    BFrag a0, a1;
    const int ar0 = (wm * 32 + ln) * 32 + 8 * hh;
    a0.q[0] = *(const uv4*)&As[ar0];
    a0.q[1] = *(const uv4*)&As[ar0 + 16];
    const int ar1 = ar0 + 16 * 32;
    a1.q[0] = *(const uv4*)&As[ar1];
    a1.q[1] = *(const uv4*)&As[ar1 + 16];
    // B lane layout: K(i) = 16*(i>>3) + 8*hh + (i&7)
    BFrag b[4];
#pragma unroll
    for (int j = 0; j < 4; ++j) {
      const int br = (nloc + 16 * j + ln) * 32 + 8 * hh;
      b[j].q[0] = *(const uv4*)&Bs[br];
      b[j].q[1] = *(const uv4*)&Bs[br + 16];
    }

#pragma unroll
    for (int j = 0; j < 4; ++j) {
      acc[0][j] = wmma_bf16(a0, b[j], acc[0][j]);
      acc[1][j] = wmma_bf16(a1, b[j], acc[1][j]);
    }

    __syncthreads();   // before next iteration overwrites As/Bs
  }

  // ---- store: C 16x16 f32 layout: VGPR r -> row (r + 8*hh), col = ln ----
  if (cMode == 0) {
    float* C = (float*)Cptr + cOffZ * z;
#pragma unroll
    for (int tm = 0; tm < 2; ++tm)
#pragma unroll
      for (int tn = 0; tn < 4; ++tn)
#pragma unroll
        for (int r = 0; r < 8; ++r)
          C[(long long)(m0 + 16 * tm + 8 * hh + r) * ldc + n0 + 16 * tn + ln] =
              acc[tm][tn][r] * alpha;
  } else if (cMode == 1) {
    u16* C = (u16*)Cptr + cOffZ * z;
#pragma unroll
    for (int tm = 0; tm < 2; ++tm)
#pragma unroll
      for (int tn = 0; tn < 4; ++tn)
#pragma unroll
        for (int r = 0; r < 8; ++r)
          C[(long long)(m0 + 16 * tm + 8 * hh + r) * ldc + n0 + 16 * tn + ln] =
              f32_to_bf16(acc[tm][tn][r] * alpha);
  } else {
    u16* C = (u16*)Cptr + cOffZ * z;
#pragma unroll
    for (int tm = 0; tm < 2; ++tm)
#pragma unroll
      for (int tn = 0; tn < 4; ++tn)
#pragma unroll
        for (int r = 0; r < 8; ++r)
          C[(long long)(n0 + 16 * tn + ln) * ldc + (m0 + 16 * tm + 8 * hh + r)] =
              f32_to_bf16(acc[tm][tn][r] * alpha);
  }
}

// dst (bf16) = src (f32), elementwise
__global__ __launch_bounds__(256) void cast_bf16(
    const float* __restrict__ src, u16* __restrict__ dst, int n)
{
  int idx = blockIdx.x * blockDim.x + threadIdx.x;
  if (idx < n) dst[idx] = f32_to_bf16(src[idx]);
}

// Wt[n, m] (bf16, [cols, rows]) = W[m, n] (f32, [rows, cols])
__global__ __launch_bounds__(256) void transpose_cast_bf16(
    const float* __restrict__ W, u16* __restrict__ Wt, int rows, int cols)
{
  int idx = blockIdx.x * blockDim.x + threadIdx.x;
  if (idx >= rows * cols) return;
  int n = idx / rows;
  int m = idx - n * rows;
  Wt[idx] = f32_to_bf16(W[(long long)m * cols + n]);
}

// In-place row softmax over 4096-wide rows; one 256-thread block per row.
__global__ __launch_bounds__(256) void softmax_rows_4096(float* __restrict__ attn)
{
  const int NIT = 16;
  float* p = attn + (long long)blockIdx.x * 4096;
  __shared__ float red[256];
  const int t = threadIdx.x;

  float vals[NIT];
  float mx = -3.402823466e38f;
#pragma unroll
  for (int i = 0; i < NIT; ++i) {
    vals[i] = p[t + i * 256];
    mx = fmaxf(mx, vals[i]);
  }
  red[t] = mx;
  __syncthreads();
  for (int s = 128; s > 0; s >>= 1) {
    if (t < s) red[t] = fmaxf(red[t], red[t + s]);
    __syncthreads();
  }
  mx = red[0];
  __syncthreads();

  float sum = 0.f;
#pragma unroll
  for (int i = 0; i < NIT; ++i) {
    vals[i] = __expf(vals[i] - mx);
    sum += vals[i];
  }
  red[t] = sum;
  __syncthreads();
  for (int s = 128; s > 0; s >>= 1) {
    if (t < s) red[t] += red[t + s];
    __syncthreads();
  }
  const float inv = 1.f / red[0];
#pragma unroll
  for (int i = 0; i < NIT; ++i)
    p[t + i * 256] = vals[i] * inv;
}

extern "C" void kernel_launch(void* const* d_in, const int* in_sizes, int n_in,
                              void* d_out, int out_size, void* d_ws, size_t ws_size,
                              hipStream_t stream) {
  (void)in_sizes; (void)n_in; (void)out_size; (void)ws_size;
  const int S = 4096, F = 1024, H = 1024, NH = 8, DK = 128;

  const float* x  = (const float*)d_in[0];
  const float* Wq = (const float*)d_in[1];
  const float* Wk = (const float*)d_in[2];
  const float* Wv = (const float*)d_in[3];
  const float* Wo = (const float*)d_in[4];

  float* outY = (float*)d_out;                       // [S, H] f32
  float* outA = outY + (size_t)S * H;                // [NH, S, S] f32

  // workspace (u16 elems): 4 weights (1M) + Q,K,Vt,Y (4M each) + xb (4M) = 48 MB
  u16* Wqt = (u16*)d_ws;
  u16* Wkt = Wqt + (size_t)1024 * 1024;
  u16* Wvt = Wkt + (size_t)1024 * 1024;
  u16* Wot = Wvt + (size_t)1024 * 1024;
  u16* Qb  = Wot + (size_t)1024 * 1024;
  u16* Kb  = Qb  + (size_t)S * H;
  u16* Vt  = Kb  + (size_t)S * H;   // [H, S] (pre-transposed V)
  u16* Yb  = Vt  + (size_t)S * H;   // [S, H]
  u16* xb  = Yb  + (size_t)S * H;   // [S, F] bf16 copy of x

  dim3 blk(256);

  // 0) x -> bf16 once; weights -> bf16, transposed to [N, K]
  cast_bf16<<<dim3((S * F) / 256), blk, 0, stream>>>(x, xb, S * F);
  transpose_cast_bf16<<<dim3((F * H) / 256), blk, 0, stream>>>(Wq, Wqt, F, H);
  transpose_cast_bf16<<<dim3((F * H) / 256), blk, 0, stream>>>(Wk, Wkt, F, H);
  transpose_cast_bf16<<<dim3((F * H) / 256), blk, 0, stream>>>(Wv, Wvt, F, H);
  transpose_cast_bf16<<<dim3((H * F) / 256), blk, 0, stream>>>(Wo, Wot, H, F);

  // 1) Q = x@Wq, K = x@Wk (bf16), V = x@Wv stored transposed (Vt[H,S])
  gemm_wmma_bf16<<<dim3(H / 128, S / 128, 1), blk, 0, stream>>>(
      xb, F, 0LL, 0, Wqt, F, 0LL, Qb, H, 0LL, 1, F, 1.0f);
  gemm_wmma_bf16<<<dim3(H / 128, S / 128, 1), blk, 0, stream>>>(
      xb, F, 0LL, 0, Wkt, F, 0LL, Kb, H, 0LL, 1, F, 1.0f);
  gemm_wmma_bf16<<<dim3(H / 128, S / 128, 1), blk, 0, stream>>>(
      xb, F, 0LL, 0, Wvt, F, 0LL, Vt, S, 0LL, 2, F, 1.0f);

  // 2) scores[h] = Q_h @ K_h^T / sqrt(dk)  -> f32 directly into attn output
  gemm_wmma_bf16<<<dim3(S / 128, S / 128, NH), blk, 0, stream>>>(
      Qb, H, (long long)DK, 0, Kb, H, (long long)DK,
      outA, S, (long long)S * S, 0, DK, 0.08838834764831843f);

  // 3) softmax rows in place
  softmax_rows_4096<<<dim3(NH * S), blk, 0, stream>>>(outA);

  // 4) Y[:, h*dk:(h+1)*dk] = attn[h] @ V_h   (A read as f32, converted in LDS)
  gemm_wmma_bf16<<<dim3(DK / 128, S / 128, NH), blk, 0, stream>>>(
      outA, S, (long long)S * S, 1, Vt, S, (long long)DK * S,
      Yb, H, (long long)DK, 1, S, 1.0f);

  // 5) y = Y @ Wo -> f32 output
  gemm_wmma_bf16<<<dim3(F / 128, S / 128, 1), blk, 0, stream>>>(
      Yb, H, 0LL, 0, Wot, H, 0LL, outY, F, 0LL, 0, H, 1.0f);
}
